// Atom91Decoder_47236050321505
// MI455X (gfx1250) — matrix operand order repaired and checked by reading, not compiled, not run on hardware
//
#include <hip/hip_runtime.h>
#include <hip/hip_bf16.h>

// ---------------- problem constants ----------------
#define NN 10000
#define KK 30
#define EE (NN * KK)   // 300000, divisible by 16
#define HH 32
#define NLAYER 4
#define FRAGS_PER_LAYER 30

typedef __attribute__((ext_vector_type(16))) __bf16 v16bf;
typedef __attribute__((ext_vector_type(8)))  __bf16 v8bf;
typedef __attribute__((ext_vector_type(2)))  __bf16 v2bf;
typedef __attribute__((ext_vector_type(8)))  float  v8f;

#define DEVINL static __device__ __forceinline__

// Hardware f32 -> bf16 (RNE): backend emits v_cvt_pk_bf16_f32 for pairs.
DEVINL __bf16 f2bf(float f) { return static_cast<__bf16>(f); }

DEVINL void store2bf(__bf16* d, float x, float y) {
  v2bf t; t[0] = f2bf(x); t[1] = f2bf(y);
  *(v2bf*)d = t;
}

// A-frag from a bf16 LDS tile (row stride ldb elements, ldb % 8 == 0, base 16B-aligned).
// lane L: row M=L&15, chunks at K = kb..kb+7 and 16+kb..16+kb+7, kb=(L>=16)*8
// -> two 16B ds_load_b128 per lane, no conversion.
DEVINL v16bf loadAb(const __bf16* t, int ldb) {
  int lane = threadIdx.x & 31;
  const __bf16* p = t + (lane & 15) * ldb + ((lane >> 4) << 3);
  v8bf lo = *(const v8bf*)p;
  v8bf hi = *(const v8bf*)(p + 16);
  v16bf a;
#pragma unroll
  for (int e = 0; e < 8; ++e) { a[e] = lo[e]; a[8 + e] = hi[e]; }
  return a;
}

// B-frag built from row-major f32 weights W[K][ldn] (used once, in prep kernel).
// lane L holds column N = n0 + (L&15); K = koff + (L>=16)*8 + e + (e>=8)*8
DEVINL v16bf loadB(const float* W, int ldn, int koff, int n0) {
  int lane = threadIdx.x & 31;
  const float* p = W + (koff + ((lane >> 4) << 3)) * ldn + n0 + (lane & 15);
  v16bf b;
#pragma unroll
  for (int e = 0; e < 16; ++e) b[e] = f2bf(p[(e + ((e >> 3) << 3)) * ldn]);
  return b;
}

// Pre-swizzled B-frag fetch: one coalesced 32B load per lane.
DEVINL v16bf loadBpre(const v16bf* __restrict__ wf, int fid) {
  return wf[(size_t)fid * 32 + (threadIdx.x & 31)];
}

DEVINL v8f wmma_bf16(v16bf a, v16bf b, v8f c) {
  return __builtin_amdgcn_wmma_f32_16x16x32_bf16(false, a, false, b, (short)0, c,
                                                 false, false);
}
// D-frag mapping: element r of lane L -> (M = r + 8*(L>>4), N = n0 + (L&15))

DEVINL float finitef(float x) { return __builtin_isfinite(x) ? x : 0.0f; }

DEVINL void norm3(float& x, float& y, float& z) {
  float n = sqrtf(x * x + y * y + z * z) + 1e-8f;
  x /= n; y /= n; z /= n;
}
DEVINL void cross3(const float* a, const float* b, float* c) {
  c[0] = a[1] * b[2] - a[2] * b[1];
  c[1] = a[2] * b[0] - a[0] * b[2];
  c[2] = a[0] * b[1] - a[1] * b[0];
}

// ============================================================
// 0a) Pre-swizzle all weight B-fragments into WMMA lane layout (bf16).
// frag layout per layer (30): [0..1] We1 | [2..3] Wss | [4..5] Wds | [6..7] We2
// [8..9] Wvs | [10..11] W0 | [12..13] W1 | [14+ko*4+ct] Wu1 | [26+ko*2+ct] Wu2
// ============================================================
__global__ __launch_bounds__(32)
void prep_frags_kernel(const float* __restrict__ We1s, const float* __restrict__ Wss,
                       const float* __restrict__ Wds, const float* __restrict__ We2s,
                       const float* __restrict__ Wvs, const float* __restrict__ W0s,
                       const float* __restrict__ W1s, const float* __restrict__ Wu1s,
                       const float* __restrict__ Wu2s, v16bf* __restrict__ wfrag) {
  int fid = blockIdx.x;
  int l = fid / FRAGS_PER_LAYER, loc = fid % FRAGS_PER_LAYER;
  const float* W;
  int ldn, koff, n0;
  if (loc < 10) {
    const float* bases[5] = {We1s, Wss, Wds, We2s, Wvs};
    W = bases[loc >> 1] + (size_t)l * 32 * 32;
    ldn = 32; koff = 0; n0 = (loc & 1) * 16;
  } else if (loc < 14) {
    W = ((loc - 10) < 2 ? W0s : W1s) + (size_t)l * 32 * 32;
    ldn = 32; koff = 0; n0 = ((loc - 10) & 1) * 16;
  } else if (loc < 26) {
    int t = loc - 14;
    W = Wu1s + (size_t)l * 96 * 64;
    ldn = 64; koff = (t >> 2) * 32; n0 = (t & 3) * 16;
  } else {
    int t = loc - 26;
    W = Wu2s + (size_t)l * 64 * 32;
    ldn = 32; koff = (t >> 1) * 32; n0 = (t & 1) * 16;
  }
  v16bf b = loadB(W, ldn, koff, n0);
  wfrag[(size_t)fid * 32 + (threadIdx.x & 31)] = b;
}

// ============================================================
// 0b) Pack CA position + batch id (knn inner loop reads one float4)
// ============================================================
__global__ void pack_pos_kernel(const float* __restrict__ bb,
                                const unsigned char* __restrict__ x_mask,
                                const int* __restrict__ batch_ids,
                                float4* __restrict__ pos) {
  int i = blockIdx.x * blockDim.x + threadIdx.x;
  if (i >= NN) return;
  bool m = x_mask[i] != 0;
  float4 p;
  p.x = m ? 1.0e6f : bb[i * 12 + 3 + 0];
  p.y = m ? 1.0e6f : bb[i * 12 + 3 + 1];
  p.z = m ? 1.0e6f : bb[i * 12 + 3 + 2];
  p.w = __int_as_float(batch_ids[i]);
  pos[i] = p;
}

// ============================================================
// 1) Node embedding
// ============================================================
__global__ void node_embed_kernel(const float* __restrict__ bb,
                                  const float* __restrict__ latent,
                                  const float* __restrict__ Wemb0,
                                  const float* __restrict__ Wemb1,
                                  float* __restrict__ h0, float* __restrict__ h1) {
  int t = threadIdx.x;
  int n = blockIdx.x * 8 + (t >> 5);
  int c = t & 31;
  if (n >= NN) return;

  float dih[3];
#pragma unroll
  for (int tt = 0; tt < 3; ++tt) {
    int j = 3 * n + tt - 1;
    float D = 0.0f;
    if (j >= 0 && j <= 3 * NN - 4) {
      float X[4][3];
#pragma unroll
      for (int m = 0; m < 4; ++m) {
        int g = j + m;
#pragma unroll
        for (int d = 0; d < 3; ++d) X[m][d] = bb[(g / 3) * 12 + (g % 3) * 3 + d];
      }
      float u2[3], u1[3], u0[3];
#pragma unroll
      for (int d = 0; d < 3; ++d) {
        u2[d] = X[1][d] - X[0][d];
        u1[d] = X[2][d] - X[1][d];
        u0[d] = X[3][d] - X[2][d];
      }
      norm3(u2[0], u2[1], u2[2]);
      norm3(u1[0], u1[1], u1[2]);
      norm3(u0[0], u0[1], u0[2]);
      float n2[3], n1[3];
      cross3(u2, u1, n2); norm3(n2[0], n2[1], n2[2]);
      cross3(u1, u0, n1); norm3(n1[0], n1[1], n1[2]);
      float cosD = n2[0] * n1[0] + n2[1] * n1[1] + n2[2] * n1[2];
      cosD = fminf(fmaxf(cosD, -1.0f + 1e-7f), 1.0f - 1e-7f);
      float sg = u2[0] * n1[0] + u2[1] * n1[1] + u2[2] * n1[2];
      float s = (sg > 0.0f) ? 1.0f : ((sg < 0.0f) ? -1.0f : 0.0f);
      D = s * acosf(cosD);
    }
    dih[tt] = D;
  }
  float df[7] = {cosf(dih[0]), cosf(dih[1]), cosf(dih[2]),
                 sinf(dih[0]), sinf(dih[1]), sinf(dih[2]), 1.0f};

  {
    float acc = 0.0f;
#pragma unroll
    for (int k = 0; k < 7; ++k) acc += df[k] * Wemb0[k * HH + c];
    for (int k = 0; k < HH; ++k)
      acc += latent[n * 4 * HH + 0 * HH + k] * Wemb0[(7 + k) * HH + c];
    h0[n * HH + c] = fmaxf(acc, 0.0f);
  }

  float ca[3], l1[7][3];
#pragma unroll
  for (int d = 0; d < 3; ++d) ca[d] = bb[n * 12 + 3 + d];
#pragma unroll
  for (int a = 0; a < 4; ++a)
#pragma unroll
    for (int d = 0; d < 3; ++d) l1[a][d] = bb[n * 12 + a * 3 + d] - ca[d];
  {
    float f[3] = {0, 0, 0};
    if (n < NN - 1) {
#pragma unroll
      for (int d = 0; d < 3; ++d) f[d] = bb[(n + 1) * 12 + 3 + d] - ca[d];
      norm3(f[0], f[1], f[2]);
    }
#pragma unroll
    for (int d = 0; d < 3; ++d) l1[4][d] = f[d];
    float bwd[3] = {0, 0, 0};
    if (n > 0) {
#pragma unroll
      for (int d = 0; d < 3; ++d) bwd[d] = ca[d] - bb[(n - 1) * 12 + 3 + d];
      norm3(bwd[0], bwd[1], bwd[2]);
#pragma unroll
      for (int d = 0; d < 3; ++d) bwd[d] = -bwd[d];
    }
#pragma unroll
    for (int d = 0; d < 3; ++d) l1[5][d] = bwd[d];
    float bv[3], cv[3], av[3];
#pragma unroll
    for (int d = 0; d < 3; ++d) {
      bv[d] = ca[d] - bb[n * 12 + 0 + d];          // CA - N
      cv[d] = bb[n * 12 + 6 + d] - ca[d];          // C - CA
    }
    cross3(bv, cv, av);
#pragma unroll
    for (int d = 0; d < 3; ++d)
      l1[6][d] = -0.58273431f * av[d] + 0.56802827f * bv[d] - 0.54067466f * cv[d];
  }
#pragma unroll
  for (int k = 0; k < 7; ++k)
#pragma unroll
    for (int d = 0; d < 3; ++d) l1[k][d] = finitef(l1[k][d]);

#pragma unroll
  for (int i = 0; i < 3; ++i) {
    float acc = 0.0f;
#pragma unroll
    for (int k = 0; k < 7; ++k) acc += l1[k][i] * Wemb1[k * HH + c];
    for (int k = 0; k < HH; ++k)
      acc += latent[n * 4 * HH + (1 + i) * HH + k] * Wemb1[(7 + k) * HH + c];
    h1[n * 3 * HH + i * HH + c] = acc;
  }
}

// ============================================================
// 2) kNN: thread-per-row scan keeping the 30 smallest d^2
// ============================================================
__global__ void knn_kernel(const float4* __restrict__ pos, int* __restrict__ src) {
  int i = blockIdx.x * blockDim.x + threadIdx.x;
  if (i >= NN) return;
  float4 pi = pos[i];
  int bi = __float_as_int(pi.w);

  float bd[KK];
  int bj[KK];
#pragma unroll
  for (int k = 0; k < KK; ++k) { bd[k] = 3.0e38f; bj[k] = 0; }
  int wpos = 0;

  for (int j = 0; j < NN; ++j) {
    if (j == i) continue;
    float4 pj = pos[j];
    float cand;
    if (__float_as_int(pj.w) != bi) {
      cand = 1.0e18f;
    } else {
      float dx = pi.x - pj.x, dy = pi.y - pj.y, dz = pi.z - pj.z;
      cand = dx * dx + dy * dy + dz * dz;
    }
    if (cand < bd[wpos]) {
      bd[wpos] = cand;
      bj[wpos] = j;
      float w = -1.0f; int wp = 0;
#pragma unroll
      for (int k = 0; k < KK; ++k)
        if (bd[k] > w) { w = bd[k]; wp = k; }
      wpos = wp;
    }
  }
#pragma unroll
  for (int k = 0; k < KK; ++k) src[i * KK + k] = bj[k];
}

// ============================================================
// 3) Edge features: ef = [rbf(16) | posemb(16)], R = edge rotation (3x3)
// ============================================================
__global__ void edge_feat_kernel(const float* __restrict__ bb,
                                 const int* __restrict__ src,
                                 float* __restrict__ ef, float* __restrict__ Rm) {
  int e = blockIdx.x * blockDim.x + threadIdx.x;
  if (e >= EE) return;
  int dsti = e / KK;
  int s = src[e];
  float v[3];
#pragma unroll
  for (int d = 0; d < 3; ++d) v[d] = bb[dsti * 12 + 3 + d] - bb[s * 12 + 3 + d];
  float dist = sqrtf(v[0] * v[0] + v[1] * v[1] + v[2] * v[2]);

  const float sig = 20.0f / 16.0f;
#pragma unroll
  for (int m = 0; m < 16; ++m) {
    float mu = 20.0f * (float)m / 15.0f;
    float t = (dist - mu) / sig;
    ef[e * 32 + m] = expf(-t * t);
  }
  float diff = (float)(s - dsti);
#pragma unroll
  for (int q = 0; q < 8; ++q) {
    float freq = expf(-((float)(2 * q) / 16.0f) * 9.210340371976184f); // ln(1e4)
    float ang = diff * freq;
    ef[e * 32 + 16 + q] = cosf(ang);
    ef[e * 32 + 24 + q] = sinf(ang);
  }

  float z[3] = {v[0] / (dist + 1e-8f), v[1] / (dist + 1e-8f), v[2] / (dist + 1e-8f)};
  float a[3];
  if (fabsf(z[0]) < 0.9f) { a[0] = 1; a[1] = 0; a[2] = 0; }
  else { a[0] = 0; a[1] = 1; a[2] = 0; }
  float x[3], y[3];
  cross3(a, z, x); norm3(x[0], x[1], x[2]);
  cross3(z, x, y);
#pragma unroll
  for (int d = 0; d < 3; ++d) {
    Rm[e * 9 + 0 + d] = x[d];
    Rm[e * 9 + 3 + d] = y[d];
    Rm[e * 9 + 6 + d] = z[d];
  }
}

// ============================================================
// 4a) Edge message kernel (one wave = 16 edges; WMMA-heavy, bf16 LDS tiles)
// ============================================================
__global__ __launch_bounds__(32)
void edge_msg_kernel(const int* __restrict__ src, const float* __restrict__ ef,
                     const float* __restrict__ h0, const float* __restrict__ h1,
                     const float* __restrict__ Rm,
                     const v16bf* __restrict__ wf, int fb,
                     float* __restrict__ a0, float* __restrict__ a1) {
  __shared__ __align__(16) __bf16 sEfB[16][40], sH0sB[16][40], sH0dB[16][40];
  __shared__ __align__(16) __bf16 sPreB[16][40], sVB[48][40];
  __shared__ float sH0sF[16][33], sG[16][33], sVW[48][33];
  __shared__ int sSrc[16], sDst[16];

  int tid = threadIdx.x;
  int lane = tid & 31;
  int e0 = blockIdx.x * 16;

  if (tid < 16) {
    int e = e0 + tid;
    sSrc[tid] = src[e];
    sDst[tid] = e / KK;
  }
  __syncthreads();

  if (e0 + 16 < EE) __builtin_prefetch(&ef[(size_t)(e0 + 16) * 32], 0, 1);

  // stage ef/h0s/h0d as bf16 (pairwise: float2 load -> cvt_pk -> b32 LDS store)
  for (int i = tid; i < 16 * 16; i += 32) {
    int r = i >> 4, c0 = (i & 15) * 2;
    float2 ev = *(const float2*)&ef[(size_t)(e0 + r) * 32 + c0];
    store2bf(&sEfB[r][c0], ev.x, ev.y);
    float2 hs = *(const float2*)&h0[(size_t)sSrc[r] * 32 + c0];
    store2bf(&sH0sB[r][c0], hs.x, hs.y);
    sH0sF[r][c0] = hs.x; sH0sF[r][c0 + 1] = hs.y;
    float2 hd = *(const float2*)&h0[(size_t)sDst[r] * 32 + c0];
    store2bf(&sH0dB[r][c0], hd.x, hd.y);
  }
  __syncthreads();

  v16bf aEf = loadAb(&sEfB[0][0], 40);
  v16bf aS  = loadAb(&sH0sB[0][0], 40);
  v16bf aD  = loadAb(&sH0dB[0][0], 40);

  // pre_g = relu(ef@We1 + h0s@Wss + h0d@Wds)
#pragma unroll
  for (int ct = 0; ct < 2; ++ct) {
    v8f acc = {};
    acc = wmma_bf16(aEf, loadBpre(wf, fb + 0 + ct), acc);
    acc = wmma_bf16(aS,  loadBpre(wf, fb + 2 + ct), acc);
    acc = wmma_bf16(aD,  loadBpre(wf, fb + 4 + ct), acc);
#pragma unroll
    for (int r = 0; r < 8; ++r) {
      int M = r + ((lane >> 4) << 3);
      int Nc = ct * 16 + (lane & 15);
      sPreB[M][Nc] = f2bf(fmaxf(acc[r], 0.0f));
    }
  }
  __syncthreads();

  // g = pre_g @ We2 ; m0 = g * h0[src] -> a0[dst]
  v16bf aP = loadAb(&sPreB[0][0], 40);
#pragma unroll
  for (int ct = 0; ct < 2; ++ct) {
    v8f g = {};
    g = wmma_bf16(aP, loadBpre(wf, fb + 6 + ct), g);
#pragma unroll
    for (int r = 0; r < 8; ++r) {
      int M = r + ((lane >> 4) << 3);
      int Nc = ct * 16 + (lane & 15);
      float gv = g[r];
      sG[M][Nc] = gv;
      atomicAdd(&a0[(size_t)sDst[M] * 32 + Nc], gv * sH0sF[M][Nc]);
    }
  }
  __syncthreads();

  // v1 = R @ h1[src]   (48 rows = edge*3+comp, 32 channels, pairwise)
  for (int i = tid; i < 48 * 16; i += 32) {
    int row = i >> 4, c0 = (i & 15) * 2;
    int ed = row / 3, comp = row % 3;
    int s = sSrc[ed];
    float ax = 0.0f, ay = 0.0f;
#pragma unroll
    for (int j = 0; j < 3; ++j) {
      float r = Rm[(size_t)(e0 + ed) * 9 + comp * 3 + j];
      float2 hv = *(const float2*)&h1[(size_t)s * 96 + j * 32 + c0];
      ax += r * hv.x; ay += r * hv.y;
    }
    store2bf(&sVB[row][c0], ax, ay);
  }
  __syncthreads();

  // vW = (v1 @ Wvs) * g
#pragma unroll
  for (int rt = 0; rt < 3; ++rt) {
    v16bf aV = loadAb(&sVB[rt * 16][0], 40);
#pragma unroll
    for (int ct = 0; ct < 2; ++ct) {
      v8f vw = {};
      vw = wmma_bf16(aV, loadBpre(wf, fb + 8 + ct), vw);
#pragma unroll
      for (int r = 0; r < 8; ++r) {
        int M = r + ((lane >> 4) << 3);
        int Nc = ct * 16 + (lane & 15);
        int rowAbs = rt * 16 + M;
        sVW[rowAbs][Nc] = vw[r] * sG[rowAbs / 3][Nc];
      }
    }
  }
  __syncthreads();

  // v2 = R^T @ vW -> a1[dst]  (pairwise)
  for (int i = tid; i < 48 * 16; i += 32) {
    int row = i >> 4, c0 = (i & 15) * 2;
    int ed = row / 3, comp = row % 3;
    float ax = 0.0f, ay = 0.0f;
#pragma unroll
    for (int j = 0; j < 3; ++j) {
      float r = Rm[(size_t)(e0 + ed) * 9 + j * 3 + comp];
      ax += r * sVW[ed * 3 + j][c0];
      ay += r * sVW[ed * 3 + j][c0 + 1];
    }
    float* dst = &a1[(size_t)sDst[ed] * 96 + comp * 32 + c0];
    atomicAdd(dst, ax);
    atomicAdd(dst + 1, ay);
  }
}

// ============================================================
// 4b) Node update: h0 += relu(a0/K @ W0); h1 += a1/K @ W1; zero accumulators
// ============================================================
__global__ __launch_bounds__(32)
void node_update_kernel(const v16bf* __restrict__ wf, int fb,
                        float* __restrict__ h0, float* __restrict__ h1,
                        float* __restrict__ a0, float* __restrict__ a1) {
  __shared__ __align__(16) __bf16 sA0B[16][40], sA1B[48][40];
  int tid = threadIdx.x;
  int lane = tid & 31;
  int n0 = blockIdx.x * 16;
  const float invK = 1.0f / (float)KK;
  const float2 fzero = {0.0f, 0.0f};

  for (int i = tid; i < 16 * 16; i += 32) {
    int r = i >> 4, c0 = (i & 15) * 2;
    size_t idx = (size_t)(n0 + r) * 32 + c0;
    float2 av = *(const float2*)&a0[idx];
    store2bf(&sA0B[r][c0], av.x * invK, av.y * invK);
    *(float2*)&a0[idx] = fzero;
  }
  for (int i = tid; i < 48 * 16; i += 32) {
    int row = i >> 4, c0 = (i & 15) * 2;
    int nd = row / 3, comp = row % 3;
    size_t idx = (size_t)(n0 + nd) * 96 + comp * 32 + c0;
    float2 av = *(const float2*)&a1[idx];
    store2bf(&sA1B[row][c0], av.x * invK, av.y * invK);
    *(float2*)&a1[idx] = fzero;
  }
  __syncthreads();

  v16bf aF = loadAb(&sA0B[0][0], 40);
#pragma unroll
  for (int ct = 0; ct < 2; ++ct) {
    v8f d = {};
    d = wmma_bf16(aF, loadBpre(wf, fb + 10 + ct), d);
#pragma unroll
    for (int r = 0; r < 8; ++r) {
      int M = r + ((lane >> 4) << 3);
      int Nc = ct * 16 + (lane & 15);
      h0[(size_t)(n0 + M) * 32 + Nc] += fmaxf(d[r], 0.0f);
    }
  }
#pragma unroll
  for (int rt = 0; rt < 3; ++rt) {
    v16bf aV = loadAb(&sA1B[rt * 16][0], 40);
#pragma unroll
    for (int ct = 0; ct < 2; ++ct) {
      v8f d = {};
      d = wmma_bf16(aV, loadBpre(wf, fb + 12 + ct), d);
#pragma unroll
      for (int r = 0; r < 8; ++r) {
        int M = r + ((lane >> 4) << 3);
        int Nc = ct * 16 + (lane & 15);
        int rowAbs = rt * 16 + M;
        int nd = rowAbs / 3, comp = rowAbs % 3;
        h1[(size_t)(n0 + nd) * 96 + comp * 32 + Nc] += d[r];
      }
    }
  }
}

// ============================================================
// 4c) Edge-feature update: ef += relu([ef|h0s|h0d] @ Wu1) @ Wu2
// ============================================================
__global__ __launch_bounds__(32)
void ef_update_kernel(const int* __restrict__ src, float* __restrict__ ef,
                      const float* __restrict__ h0,
                      const v16bf* __restrict__ wf, int fb) {
  __shared__ __align__(16) __bf16 sCatB[16][104], sMidB[16][72];
  __shared__ int sSrc[16], sDst[16];
  int tid = threadIdx.x;
  int lane = tid & 31;
  int e0 = blockIdx.x * 16;

  if (tid < 16) {
    int e = e0 + tid;
    sSrc[tid] = src[e];
    sDst[tid] = e / KK;
  }
  __syncthreads();

  // stage [ef | h0s | h0d] pairs (region boundaries at 32/64 are even -> no straddle)
  for (int i = tid; i < 16 * 48; i += 32) {
    int r = i / 48, c0 = (i % 48) * 2;
    float2 v;
    if (c0 < 32)       v = *(const float2*)&ef[(size_t)(e0 + r) * 32 + c0];
    else if (c0 < 64)  v = *(const float2*)&h0[(size_t)sSrc[r] * 32 + (c0 - 32)];
    else               v = *(const float2*)&h0[(size_t)sDst[r] * 32 + (c0 - 64)];
    store2bf(&sCatB[r][c0], v.x, v.y);
  }
  __syncthreads();

  v16bf aC0 = loadAb(&sCatB[0][0],  104);
  v16bf aC1 = loadAb(&sCatB[0][32], 104);
  v16bf aC2 = loadAb(&sCatB[0][64], 104);

#pragma unroll
  for (int ct = 0; ct < 4; ++ct) {
    v8f m = {};
    m = wmma_bf16(aC0, loadBpre(wf, fb + 14 + 0 * 4 + ct), m);
    m = wmma_bf16(aC1, loadBpre(wf, fb + 14 + 1 * 4 + ct), m);
    m = wmma_bf16(aC2, loadBpre(wf, fb + 14 + 2 * 4 + ct), m);
#pragma unroll
    for (int r = 0; r < 8; ++r) {
      int M = r + ((lane >> 4) << 3);
      int Nc = ct * 16 + (lane & 15);
      sMidB[M][Nc] = f2bf(fmaxf(m[r], 0.0f));
    }
  }
  __syncthreads();

  v16bf aM0 = loadAb(&sMidB[0][0],  72);
  v16bf aM1 = loadAb(&sMidB[0][32], 72);
#pragma unroll
  for (int ct = 0; ct < 2; ++ct) {
    v8f o = {};
    o = wmma_bf16(aM0, loadBpre(wf, fb + 26 + 0 * 2 + ct), o);
    o = wmma_bf16(aM1, loadBpre(wf, fb + 26 + 1 * 2 + ct), o);
#pragma unroll
    for (int r = 0; r < 8; ++r) {
      int M = r + ((lane >> 4) << 3);
      int Nc = ct * 16 + (lane & 15);
      ef[(size_t)(e0 + M) * 32 + Nc] += o[r];
    }
  }
}

// ============================================================
// 5) Head: atom91 (= rot @ local, trans cancels) + seq log-softmax
// ============================================================
__device__ __constant__ int cAA[87] = {
  0, 1,1,1,1,1,1,1, 2,2,2,2, 3,3,3,3, 4,4, 5,5,5,5,5, 6,6,6,6,6,
  8,8,8,8,8,8, 9,9,9,9, 10,10,10,10, 11,11,11,11,11, 12,12,12,12,
  13,13,13,13,13,13,13, 14,14,14, 15,15, 16,16,16,
  17,17,17,17,17,17,17,17,17,17, 18,18,18,18,18,18,18,18, 19,19,19};
__device__ __constant__ int cOFF[87] = {
  0, 0,1,2,3,4,5,6, 0,1,2,3, 0,1,2,3, 0,1, 0,1,2,3,4, 0,1,2,3,4,
  0,1,2,3,4,5, 0,1,2,3, 0,1,2,3, 0,1,2,3,4, 0,1,2,3,
  0,1,2,3,4,5,6, 0,1,2, 0,1, 0,1,2,
  0,1,2,3,4,5,6,7,8,9, 0,1,2,3,4,5,6,7, 0,1,2};

__global__ __launch_bounds__(128)
void head_kernel(const float* __restrict__ h0, const float* __restrict__ rot,
                 const float* __restrict__ Wp0, const float* __restrict__ Wt,
                 const float* __restrict__ bt,
                 const float* __restrict__ Wseq1, const float* __restrict__ bseq1,
                 const float* __restrict__ Wseq2, const float* __restrict__ bseq2,
                 const float* __restrict__ Wseq3, const float* __restrict__ bseq3,
                 const float* __restrict__ lit_pos, const float* __restrict__ Wchi,
                 const float* __restrict__ Wpsi,
                 float* __restrict__ out_atom, float* __restrict__ out_seq) {
  __shared__ float sH[4][32], sT0[4][32], sU[4][162], sChi[4][162];
  __shared__ float sHid1[4][64], sHid2[4][32], sLg[4][20];
  int tid = threadIdx.x;
  int w = tid >> 5, lane = tid & 31;
  int n = blockIdx.x * 4 + w;

  sH[w][lane] = h0[(size_t)n * 32 + lane];
  __syncthreads();

  {
    float acc = 0.0f;
    for (int k = 0; k < 32; ++k) acc += sH[w][k] * Wp0[k * 32 + lane];
    sT0[w][lane] = fmaxf(acc, 0.0f);
  }
  __syncthreads();

  for (int o = lane; o < 162; o += 32) {
    float acc = bt[o];
    for (int k = 0; k < 32; ++k) acc += sT0[w][k] * Wt[k * 162 + o];
    sU[w][o] = acc;
  }
  __syncthreads();

  for (int p = lane; p < 81; p += 32) {
    float x = sU[w][2 * p], y = sU[w][2 * p + 1];
    float xa = x + 1e-8f, ya = y + 1e-8f;
    float dn = sqrtf(xa * xa + ya * ya);
    sChi[w][2 * p]     = x / dn;
    sChi[w][2 * p + 1] = y / dn;
  }
  __syncthreads();

  // atoms: atom91 = rot @ local (trans cancels)
  for (int idx = lane; idx < 91; idx += 32) {
    int k, a;
    if (idx < 4) { k = 0; a = idx; }
    else { k = cAA[idx - 4]; a = 4 + cOFF[idx - 4]; }
    float loc[3];
#pragma unroll
    for (int d = 0; d < 3; ++d) {
      float v = lit_pos[(k * 14 + a) * 3 + d];
#pragma unroll
      for (int c = 0; c < 8; ++c)
        v += sChi[w][(1 + k * 4 + (c >> 1)) * 2 + (c & 1)] * Wchi[(c * 14 + a) * 3 + d];
#pragma unroll
      for (int c = 0; c < 2; ++c)
        v += sChi[w][c] * Wpsi[(c * 14 + a) * 3 + d];
      loc[d] = v;
    }
#pragma unroll
    for (int i = 0; i < 3; ++i) {
      float o = 0.0f;
#pragma unroll
      for (int j = 0; j < 3; ++j) o += rot[(size_t)n * 9 + i * 3 + j] * loc[j];
      out_atom[((size_t)n * 91 + idx) * 3 + i] = o;
    }
  }

  // sequence MLP + log-softmax
  for (int o = lane; o < 64; o += 32) {
    float acc = bseq1[o];
    for (int k = 0; k < 32; ++k) acc += sH[w][k] * Wseq1[k * 64 + o];
    sHid1[w][o] = fmaxf(acc, 0.0f);
  }
  __syncthreads();
  {
    float acc = bseq2[lane];
    for (int k = 0; k < 64; ++k) acc += sHid1[w][k] * Wseq2[k * 32 + lane];
    sHid2[w][lane] = fmaxf(acc, 0.0f);
  }
  __syncthreads();
  if (lane < 20) {
    float acc = bseq3[lane];
    for (int k = 0; k < 32; ++k) acc += sHid2[w][k] * Wseq3[k * 20 + lane];
    sLg[w][lane] = acc;
  }
  __syncthreads();
  if (lane < 20) {
    float m = -3.0e38f;
    for (int k = 0; k < 20; ++k) m = fmaxf(m, sLg[w][k]);
    float s = 0.0f;
    for (int k = 0; k < 20; ++k) s += expf(sLg[w][k] - m);
    out_seq[(size_t)n * 20 + lane] = sLg[w][lane] - m - logf(s);
  }
}

// ============================================================
// Launcher
// ============================================================
extern "C" void kernel_launch(void* const* d_in, const int* in_sizes, int n_in,
                              void* d_out, int out_size, void* d_ws, size_t ws_size,
                              hipStream_t stream) {
  (void)in_sizes; (void)n_in; (void)out_size; (void)ws_size;

  const float* bb      = (const float*)d_in[0];
  const float* latent  = (const float*)d_in[1];
  const float* rrot    = (const float*)d_in[2];
  // d_in[3] rigids_trans: cancels out in atom91 -> unused
  const float* Wemb0   = (const float*)d_in[4];
  const float* Wemb1   = (const float*)d_in[5];
  const float* We1s    = (const float*)d_in[6];
  const float* Wss     = (const float*)d_in[7];
  const float* Wds     = (const float*)d_in[8];
  const float* We2s    = (const float*)d_in[9];
  const float* Wvs     = (const float*)d_in[10];
  const float* W0s     = (const float*)d_in[11];
  const float* W1s     = (const float*)d_in[12];
  const float* Wu1s    = (const float*)d_in[13];
  const float* Wu2s    = (const float*)d_in[14];
  const float* Wp0     = (const float*)d_in[15];
  const float* Wt      = (const float*)d_in[16];
  const float* bt      = (const float*)d_in[17];
  const float* Wseq1   = (const float*)d_in[18];
  const float* bseq1   = (const float*)d_in[19];
  const float* Wseq2   = (const float*)d_in[20];
  const float* bseq2   = (const float*)d_in[21];
  const float* Wseq3   = (const float*)d_in[22];
  const float* bseq3   = (const float*)d_in[23];
  const float* lit_pos = (const float*)d_in[24];
  const float* Wchi    = (const float*)d_in[25];
  const float* Wpsi    = (const float*)d_in[26];
  const unsigned char* x_mask = (const unsigned char*)d_in[27];
  const int* batch_ids = (const int*)d_in[28];

  // workspace layout
  char* ws = (char*)d_ws;
  size_t off = 0;
  auto take = [&](size_t bytes) { char* p = ws + off; off = (off + bytes + 255) & ~(size_t)255; return p; };
  int*    src   = (int*)   take((size_t)EE * sizeof(int));
  float*  ef    = (float*) take((size_t)EE * 32 * sizeof(float));
  float*  Rm    = (float*) take((size_t)EE * 9 * sizeof(float));
  float*  h0    = (float*) take((size_t)NN * 32 * sizeof(float));
  float*  h1    = (float*) take((size_t)NN * 96 * sizeof(float));
  float*  a0    = (float*) take((size_t)NN * 32 * sizeof(float));
  float*  a1    = (float*) take((size_t)NN * 96 * sizeof(float));
  float4* pos   = (float4*)take((size_t)NN * sizeof(float4));
  v16bf*  wfrag = (v16bf*) take((size_t)NLAYER * FRAGS_PER_LAYER * 32 * sizeof(v16bf));

  hipMemsetAsync(a0, 0, (size_t)NN * 32 * sizeof(float), stream);
  hipMemsetAsync(a1, 0, (size_t)NN * 96 * sizeof(float), stream);

  prep_frags_kernel<<<NLAYER * FRAGS_PER_LAYER, 32, 0, stream>>>(
      We1s, Wss, Wds, We2s, Wvs, W0s, W1s, Wu1s, Wu2s, wfrag);
  pack_pos_kernel<<<(NN + 255) / 256, 256, 0, stream>>>(bb, x_mask, batch_ids, pos);
  node_embed_kernel<<<(NN + 7) / 8, 256, 0, stream>>>(bb, latent, Wemb0, Wemb1, h0, h1);
  knn_kernel<<<(NN + 255) / 256, 256, 0, stream>>>(pos, src);
  edge_feat_kernel<<<(EE + 255) / 256, 256, 0, stream>>>(bb, src, ef, Rm);

  for (int l = 0; l < NLAYER; ++l) {
    int fb = l * FRAGS_PER_LAYER;
    edge_msg_kernel<<<EE / 16, 32, 0, stream>>>(src, ef, h0, h1, Rm, wfrag, fb, a0, a1);
    node_update_kernel<<<NN / 16, 32, 0, stream>>>(wfrag, fb, h0, h1, a0, a1);
    ef_update_kernel<<<EE / 16, 32, 0, stream>>>(src, ef, h0, wfrag, fb);
  }

  float* out_atom = (float*)d_out;
  float* out_seq  = out_atom + (size_t)NN * 91 * 3;
  head_kernel<<<NN / 4, 128, 0, stream>>>(
      h0, rrot, Wp0, Wt, bt, Wseq1, bseq1, Wseq2, bseq2, Wseq3, bseq3,
      lit_pos, Wchi, Wpsi, out_atom, out_seq);
}
